// T5Pooler_34608846471536
// MI455X (gfx1250) — matrix operand order repaired and checked by reading, not compiled, not run on hardware
//
#include <hip/hip_runtime.h>
#include <hip/hip_bf16.h>

typedef _Float16 v16h __attribute__((ext_vector_type(16)));
typedef _Float16 v8h  __attribute__((ext_vector_type(8)));
typedef _Float16 v4h  __attribute__((ext_vector_type(4)));
typedef float    v8f  __attribute__((ext_vector_type(8)));
typedef float    v4f  __attribute__((ext_vector_type(4)));

#define N_PASS 64
#define L_SEQ  512
#define H_DIM  768
#define S_SP   16
#define BM     128
#define BN     128
#define BK     32
#define LDSROW 40   // halfs per LDS row: 32 data + 8 pad = 80 bytes (16B aligned, conflict-free)

// pooled[r, o] = tanh( sum_h X[r,h] * W[o,h] + bias[o] ),  r in [0, N*L), o in [0, H)
// outP[n, o]          = pooled[n*L + 0, o]
// outS[n*S + s, o]   += pooled[n*L + l, o] / count(s)   for l in span s of passage n
__global__ __launch_bounds__(256) void pooler_gemm_kernel(
    const float* __restrict__ X, const float* __restrict__ W,
    const float* __restrict__ bias, const int* __restrict__ spans,
    float* __restrict__ outP, float* __restrict__ outS)
{
  __shared__ _Float16 ldsA[BM * LDSROW];
  __shared__ _Float16 ldsB[BN * LDSROW];
  __shared__ int   sSpan[BM];
  __shared__ float sScale[BM];

  const int tid     = threadIdx.x;
  const int rowBase = blockIdx.y * BM;      // row into the (N*L) dimension
  const int colBase = blockIdx.x * BN;      // col into the H dimension
  const int n       = rowBase / L_SEQ;      // BM divides L -> one passage per block
  const int lBase   = rowBase % L_SEQ;

  const int* sp = spans + n * S_SP * 2;

  // Data-dependent block skip: spans are disjoint & ordered, so the covered
  // l-range is [sp[0], sp[2S-1]). Only rows with l==0 or l in that range matter.
  {
    const int s0 = sp[0];
    const int eL = sp[2 * S_SP - 1];
    const bool overlap = (lBase < eL) && (lBase + BM > s0);
    if (lBase != 0 && !overlap) return;   // uniform across block, before any barrier
  }

  // Per-row span id and 1/count (128 rows, one thread each)
  if (tid < BM) {
    const int l = lBase + tid;
    int sid = -1; float sc = 0.f;
    for (int i = 0; i < S_SP; ++i) {
      const int st = sp[2 * i], en = sp[2 * i + 1];
      if (l >= st && l < en) { sid = i; sc = 1.f / (float)(en - st); }
    }
    sSpan[tid]  = sid;
    sScale[tid] = sc;
  }

  const int  lane = tid & 31;
  const int  wid  = tid >> 5;
  const int  mwB  = (wid & 3) * 32;   // wave row base within tile (4 waves down)
  const int  nwB  = (wid >> 2) * 64;  // wave col base within tile (2 waves across)
  const bool hiL  = (lane >= 16);
  const int  l16  = lane & 15;

  v8f acc[2][4] = {};

  for (int k0 = 0; k0 < H_DIM; k0 += BK) {
    // ---- stage A (X rows) and B (W rows) into LDS as f16 ----
    // 128x32 f32 tile = 1024 float4 chunks; 256 threads x 4 chunks, fully coalesced.
    #pragma unroll
    for (int i = 0; i < 4; ++i) {
      const int q   = tid + i * 256;
      const int row = q >> 3;
      const int c4  = q & 7;
      v4f a = *(const v4f*)(X + (size_t)(rowBase + row) * H_DIM + k0 + c4 * 4);
      v4h ah = { (_Float16)a[0], (_Float16)a[1], (_Float16)a[2], (_Float16)a[3] };
      *(v4h*)(&ldsA[row * LDSROW + c4 * 4]) = ah;
      v4f b = *(const v4f*)(W + (size_t)(colBase + row) * H_DIM + k0 + c4 * 4);
      v4h bh = { (_Float16)b[0], (_Float16)b[1], (_Float16)b[2], (_Float16)b[3] };
      *(v4h*)(&ldsB[row * LDSROW + c4 * 4]) = bh;
    }
    __syncthreads();

    // ---- load fragments per the 16-bit WMMA VGPR layouts ----
    union { v16h v; v8h h[2]; } af[2], bf[4];
    #pragma unroll
    for (int mt = 0; mt < 2; ++mt) {
      // A 16x32: lane holds row (lane&15); lanes 0-15 K=[0..7]+[16..23], lanes 16-31 K=[8..15]+[24..31]
      const _Float16* ap = &ldsA[(mwB + mt * 16 + l16) * LDSROW + (hiL ? 8 : 0)];
      af[mt].h[0] = *(const v8h*)ap;
      af[mt].h[1] = *(const v8h*)(ap + 16);
    }
    #pragma unroll
    for (int nt = 0; nt < 4; ++nt) {
      // B 32x16: lane holds col (lane&15); lanes 0-15 K=[0..7]+[16..23], lanes 16-31 K=[8..15]+[24..31]
      // B[k, c] = W[colBase+c, k0+k] -> contiguous along the W row.
      const _Float16* bp = &ldsB[(nwB + nt * 16 + l16) * LDSROW + (hiL ? 8 : 0)];
      bf[nt].h[0] = *(const v8h*)bp;
      bf[nt].h[1] = *(const v8h*)(bp + 16);
    }

    #pragma unroll
    for (int mt = 0; mt < 2; ++mt)
      #pragma unroll
      for (int nt = 0; nt < 4; ++nt)
        acc[mt][nt] = __builtin_amdgcn_wmma_f32_16x16x32_f16(
            false, af[mt].v, false, bf[nt].v, (short)0, acc[mt][nt], false, false);
    __syncthreads();
  }

  // ---- fused epilogue: bias + tanh + route to outputs ----
  #pragma unroll
  for (int nt = 0; nt < 4; ++nt) {
    const int   c  = colBase + nwB + nt * 16 + l16;
    const float bv = bias[c];
    #pragma unroll
    for (int mt = 0; mt < 2; ++mt) {
      #pragma unroll
      for (int j = 0; j < 8; ++j) {
        // C/D layout: VGPR j -> M = j (lanes 0-15) or j+8 (lanes 16-31), N = lane&15
        const int   mloc = mwB + mt * 16 + j + (hiL ? 8 : 0);
        const float v    = tanhf(acc[mt][nt][j] + bv);
        const int   l    = lBase + mloc;
        if (l == 0) outP[(size_t)n * H_DIM + c] = v;
        const int sid = sSpan[mloc];
        if (sid >= 0)
          atomicAdd(outS + ((size_t)(n * S_SP + sid) * H_DIM + c), v * sScale[mloc]);
      }
    }
  }
}

__global__ void zero_kernel(float* __restrict__ p, int count) {
  const int i = blockIdx.x * blockDim.x + threadIdx.x;
  if (i < count) p[i] = 0.f;
}

extern "C" void kernel_launch(void* const* d_in, const int* in_sizes, int n_in,
                              void* d_out, int out_size, void* d_ws, size_t ws_size,
                              hipStream_t stream) {
  const float* X     = (const float*)d_in[0];   // hidden_states (N, L, H) f32
  const float* W     = (const float*)d_in[1];   // dense_w (H, H) f32
  const float* bias  = (const float*)d_in[2];   // dense_b (H,) f32
  // d_in[3] = attention_mask (all ones, unused by the reference math)
  const int*   spans = (const int*)d_in[4];     // (N, S, 2) int32

  float* out  = (float*)d_out;
  float* outP = out;                                        // (N, H)
  float* outS = out + (size_t)N_PASS * H_DIM;               // (N*S, H)
  // sentence_mask (N*S zeros) sits right after outS; zero both together.

  const int tail = N_PASS * S_SP * H_DIM + N_PASS * S_SP;   // 786432 + 1024
  zero_kernel<<<(tail + 255) / 256, 256, 0, stream>>>(outS, tail);

  dim3 grid(H_DIM / BN, (N_PASS * L_SEQ) / BM);             // (6, 256)
  pooler_gemm_kernel<<<grid, 256, 0, stream>>>(X, W, bias, spans, outP, outS);
}
